// SAM_lowRank_75325136437770
// MI455X (gfx1250) — matrix operand order repaired and checked by reading, not compiled, not run on hardware
//
#include <hip/hip_runtime.h>

// ---------------------------------------------------------------------------
// SAM low-rank attention pipeline for MI455X (gfx1250, wave32, WMMA).
// GEMMs: bf16 v_wmma_f32_16x16x32_bf16, f32 accum.
// A-tiles (pre-converted bf16 weights): GLOBAL_LOAD_ASYNC_TO_LDS_B128 staging.
// B-tiles: f32 -> bf16 transpose staging through LDS (coalesced reads).
// ---------------------------------------------------------------------------

typedef __attribute__((ext_vector_type(16))) __bf16 v16bf;
typedef __attribute__((ext_vector_type(8)))  float  v8f;

#define HW 4096                      // 64*64 pixels per plane
#define LDS_STRIDE 40                // u16 elements; 80B row pitch breaks bank conflicts
#define RSQ 0.99999500003749937f     // 1/sqrt(1+1e-5)  (the reference bn "variance")

static __device__ __forceinline__ unsigned short f2bf(float f) {
  union { __bf16 b; unsigned short u; } r;
  r.b = (__bf16)f;                   // RNE fptrunc; backend picks native cvt
  return r.u;
}
static __device__ __forceinline__ unsigned pack_bf16(float lo, float hi) {
  union { __bf16 b[2]; unsigned u; } r;
  r.b[0] = (__bf16)lo; r.b[1] = (__bf16)hi;
  return r.u;
}

static __device__ __forceinline__ void wait_async0() {
#if __has_builtin(__builtin_amdgcn_s_wait_asynccnt)
  __builtin_amdgcn_s_wait_asynccnt(0);
#else
  asm volatile("s_wait_asynccnt 0x0" ::: "memory");
#endif
}

// Async copy one 16B segment of the bf16 A-tile (64 rows x 32 k) into padded LDS.
// Per-lane LDS dest address comes from a VGPR; flat shared addr truncates to the
// LDS byte offset (ISA 10.2).
static __device__ __forceinline__ void stage_A_async(
    const unsigned short* __restrict__ Wbf, int lda,
    int mbase, int kbase, unsigned short* Al, int tid)
{
  int r = tid >> 2, seg = tid & 3;
  const unsigned short* g = Wbf + (long)(mbase + r) * lda + kbase + seg * 8;
  unsigned lds = (unsigned)(size_t)(Al + r * LDS_STRIDE + seg * 8);
  asm volatile("global_load_async_to_lds_b128 %0, %1, off"
               :: "v"(lds), "v"(g) : "memory");
}

// A-fragment (16x32 bf16): lane L (row=L&15, lhi=L>>4) holds K = [lhi*8,+8) and [16+lhi*8,+8)
static __device__ __forceinline__ v16bf ldfragA(const unsigned short* Al, int row, int lhi) {
  const unsigned short* p = Al + row * LDS_STRIDE + lhi * 8;
  union { uint4 u[2]; v16bf v; } r;
  r.u[0] = *(const uint4*)p;
  r.u[1] = *(const uint4*)(p + 16);
  return r.v;
}
// B-fragment (32x16 bf16): lane L (col=L&15, lhi=L>>4) holds K = [lhi*8,+8) and [16+lhi*8,+8)
static __device__ __forceinline__ v16bf ldfragB(const unsigned short* Bl, int col, int lhi) {
  const unsigned short* p = Bl + col * LDS_STRIDE + lhi * 8;
  union { uint4 u[2]; v16bf v; } r;
  r.u[0] = *(const uint4*)p;
  r.u[1] = *(const uint4*)(p + 16);
  return r.v;
}

// Stage one K-chunk (32) of B (128 cols, transposed) into LDS as bf16.
static __device__ __forceinline__ void stage_B(
    const float* __restrict__ B, int ldb,
    int nbase, int kbase, unsigned short* Bl, int tid)
{
  int k0 = (tid >> 4) * 2;
  int n0 = (tid & 15) * 8;
  const float* s0 = B + (long)(kbase + k0) * ldb + nbase + n0;
  const float* s1 = s0 + ldb;
  float4 a0 = ((const float4*)s0)[0], a1 = ((const float4*)s0)[1];
  float4 b0 = ((const float4*)s1)[0], b1 = ((const float4*)s1)[1];
  float va[8] = {a0.x,a0.y,a0.z,a0.w,a1.x,a1.y,a1.z,a1.w};
  float vb[8] = {b0.x,b0.y,b0.z,b0.w,b1.x,b1.y,b1.z,b1.w};
#pragma unroll
  for (int i = 0; i < 8; ++i)
    *(unsigned*)(Bl + (n0 + i) * LDS_STRIDE + k0) = pack_bf16(va[i], vb[i]);
}

// Generic C[M x N] = A[M x K] * B[K x N] (per-batch B/C). A is bf16, B is f32.
// Block tile 64(M) x 128(N); 8 waves = 4(M) x 2(N); each wave 16 x 64 via 4 WMMA accs.
__global__ __launch_bounds__(256) void k_gemm_bf16(
    const unsigned short* __restrict__ A, int lda,
    const float* __restrict__ B, int ldb, long bstrB,
    float* __restrict__ C, long bstrC, int NC)
{
  __shared__ unsigned short Al[2][64  * LDS_STRIDE];
  __shared__ unsigned short Bl[2][128 * LDS_STRIDE];
  const int tid   = threadIdx.x;
  const int nbase = blockIdx.x * 128;
  const int mbase = blockIdx.y * 64;
  const float* Bp = B + (long)blockIdx.z * bstrB;
  float*       Cp = C + (long)blockIdx.z * bstrC;

  const int lane = tid & 31, wid = tid >> 5;
  const int wm = (wid & 3) * 16, wn = (wid >> 2) * 64;
  const int lrow = lane & 15, lhi = lane >> 4;

  v8f acc[4] = {};
  stage_A_async(A, lda, mbase, 0, Al[0], tid);
  stage_B(Bp, ldb, nbase, 0, Bl[0], tid);
  wait_async0();
  __syncthreads();
  int buf = 0;
  for (int kc = 0; kc < NC; ++kc) {
    if (kc + 1 < NC) {
      stage_A_async(A, lda, mbase, (kc + 1) * 32, Al[buf ^ 1], tid);
      stage_B(Bp, ldb, nbase, (kc + 1) * 32, Bl[buf ^ 1], tid);
    }
    if (kc + 2 < NC)  // global_prefetch_b8 two chunks ahead
      __builtin_prefetch(Bp + (long)((kc + 2) * 32 + (tid >> 3)) * ldb + nbase + (tid & 7) * 16, 0, 0);
    v16bf af = ldfragA(Al[buf], wm + lrow, lhi);
#pragma unroll
    for (int j = 0; j < 4; ++j) {
      v16bf bfr = ldfragB(Bl[buf], wn + j * 16 + lrow, lhi);
      acc[j] = __builtin_amdgcn_wmma_f32_16x16x32_bf16(false, af, false, bfr,
                                                       (short)0, acc[j], false, false);
    }
    wait_async0();
    __syncthreads();
    buf ^= 1;
  }
  // D layout: element (m,n) -> vgpr m&7, lane (m>>3)*16 + n
#pragma unroll
  for (int j = 0; j < 4; ++j) {
    union { v8f v; float f[8]; } u; u.v = acc[j];
#pragma unroll
    for (int r = 0; r < 8; ++r)
      Cp[(long)(mbase + wm + lhi * 8 + r) * ldb + nbase + wn + j * 16 + lrow] = u.f[r];
  }
}

// GEMM2 B staging: 3x3 reflect-pad gather of kf + bn1 + relu, on the fly.
static __device__ __forceinline__ void stage_key_B(
    const float* __restrict__ g1, const float* __restrict__ b1,
    const float* __restrict__ Tkf, int nbase, int kbase,
    unsigned short* Bl, int tid)
{
  int k0 = (tid >> 4) * 2;
  int n0 = (tid & 15) * 8;
  float av[2][8];
#pragma unroll
  for (int kk = 0; kk < 2; ++kk) {
    int k = kbase + k0 + kk;
    int c = k / 9, p = k - c * 9;
    int di = p / 3 - 1, dj = p - (p / 3) * 3 - 1;
    float s = g1[k] * RSQ, bb = b1[k];
    const float* row = Tkf + (long)c * HW;
#pragma unroll
    for (int i = 0; i < 8; ++i) {
      int px = nbase + n0 + i;
      int h = px >> 6, w = px & 63;
      int h2 = h + di; h2 = h2 < 0 ? 1 : (h2 > 63 ? 62 : h2);
      int w2 = w + dj; w2 = w2 < 0 ? 1 : (w2 > 63 ? 62 : w2);
      float v = row[(h2 << 6) + w2];
      av[kk][i] = fmaxf(v * s + bb, 0.0f);
    }
  }
#pragma unroll
  for (int i = 0; i < 8; ++i)
    *(unsigned*)(Bl + (n0 + i) * LDS_STRIDE + k0) = pack_bf16(av[0][i], av[1][i]);
}

__global__ __launch_bounds__(256) void k_key_gemm(
    const unsigned short* __restrict__ Wk,  // bf16 w_key 64x576
    const float* __restrict__ g1, const float* __restrict__ b1,
    const float* __restrict__ T, float* __restrict__ X2)
{
  __shared__ unsigned short Al[2][64  * LDS_STRIDE];
  __shared__ unsigned short Bl[2][128 * LDS_STRIDE];
  const int tid   = threadIdx.x;
  const int nbase = blockIdx.x * 128;
  const int b     = blockIdx.z;
  const float* Tkf = T + (long)b * 256 * HW + (long)64 * HW;  // kf channels [64,128)
  float*       Xp  = X2 + (long)b * 64 * HW;

  const int lane = tid & 31, wid = tid >> 5;
  const int wm = (wid & 3) * 16, wn = (wid >> 2) * 64;
  const int lrow = lane & 15, lhi = lane >> 4;

  v8f acc[4] = {};
  stage_A_async(Wk, 576, 0, 0, Al[0], tid);
  stage_key_B(g1, b1, Tkf, nbase, 0, Bl[0], tid);
  wait_async0();
  __syncthreads();
  int buf = 0;
  const int NC = 18;  // K = 576
  for (int kc = 0; kc < NC; ++kc) {
    if (kc + 1 < NC) {
      stage_A_async(Wk, 576, 0, (kc + 1) * 32, Al[buf ^ 1], tid);
      stage_key_B(g1, b1, Tkf, nbase, (kc + 1) * 32, Bl[buf ^ 1], tid);
    }
    v16bf af = ldfragA(Al[buf], wm + lrow, lhi);
#pragma unroll
    for (int j = 0; j < 4; ++j) {
      v16bf bfr = ldfragB(Bl[buf], wn + j * 16 + lrow, lhi);
      acc[j] = __builtin_amdgcn_wmma_f32_16x16x32_bf16(false, af, false, bfr,
                                                       (short)0, acc[j], false, false);
    }
    wait_async0();
    __syncthreads();
    buf ^= 1;
  }
#pragma unroll
  for (int j = 0; j < 4; ++j) {
    union { v8f v; float f[8]; } u; u.v = acc[j];
#pragma unroll
    for (int r = 0; r < 8; ++r)
      Xp[(long)(wm + lhi * 8 + r) * HW + nbase + wn + j * 16 + lrow] = u.f[r];
  }
}

// f32 -> bf16 weight pre-conversion (layout preserved)
__global__ __launch_bounds__(256) void k_cvt_bf16(
    const float* __restrict__ src, unsigned short* __restrict__ dst, int n)
{
  int i = blockIdx.x * 256 + threadIdx.x;
  if (i < n) dst[i] = f2bf(src[i]);
}

// 4x4 avg-pool of [q ; x2] (128 ch) + bn2 + relu -> QK[b][128][256] (f32)
__global__ __launch_bounds__(256) void k_pool_bn(
    const float* __restrict__ T, const float* __restrict__ X2,
    const float* __restrict__ g2, const float* __restrict__ b2,
    float* __restrict__ QK)
{
  int id = blockIdx.x * 256 + threadIdx.x;   // 8*128*256 total
  int pq = id & 255;
  int ch = (id >> 8) & 127;
  int b  = id >> 15;
  const float* src = (ch < 64) ? (T  + ((long)b * 256 + ch)        * HW)
                               : (X2 + ((long)b * 64  + (ch - 64)) * HW);
  int hq = pq >> 4, wq = pq & 15;
  const float* p = src + (hq * 4) * 64 + wq * 4;
  float s = 0.f;
#pragma unroll
  for (int r = 0; r < 4; ++r) {
    float4 v = *(const float4*)(p + r * 64);
    s += v.x + v.y + v.z + v.w;
  }
  float m = s * (1.0f / 16.0f);
  QK[((long)b * 128 + ch) * 256 + pq] = fmaxf(m * (g2[ch] * RSQ) + b2[ch], 0.0f);
}

// out[b,o,h,w] = sum_i eh[i] * sum_j ew[j] * v[o][h+i-1][w+j-1] (zero pad)
__global__ __launch_bounds__(256) void k_apply(
    const float* __restrict__ EMB, const float* __restrict__ T, float* __restrict__ out)
{
  long id = (long)blockIdx.x * 256 + threadIdx.x;   // 8*128*4096 total
  int px = (int)(id & 4095);
  int o  = (int)((id >> 12) & 127);
  int b  = (int)(id >> 19);
  int h = px >> 6, w = px & 63;
  int pq = ((h >> 2) << 4) | (w >> 2);
  const float* Eb = EMB + (long)b * 768 * 256;
  const float* eh = Eb + (long)(o * 3) * 256 + pq;
  const float* ew = Eb + (long)(384 + o * 3) * 256 + pq;
  const float* V  = T + ((long)b * 256 + 128 + o) * HW;
  float ehv[3] = { eh[0], eh[256], eh[512] };
  float ewv[3] = { ew[0], ew[256], ew[512] };
  float acc = 0.f;
#pragma unroll
  for (int i = 0; i < 3; ++i) {
    int hh = h + i - 1;
    if (hh < 0 || hh > 63) continue;
    const float* row = V + hh * 64;
    float rs = 0.f;
#pragma unroll
    for (int j = 0; j < 3; ++j) {
      int ww = w + j - 1;
      if (ww >= 0 && ww <= 63) rs += ewv[j] * row[ww];
    }
    acc += ehv[i] * rs;
  }
  out[id] = acc;
}

extern "C" void kernel_launch(void* const* d_in, const int* in_sizes, int n_in,
                              void* d_out, int out_size, void* d_ws, size_t ws_size,
                              hipStream_t stream) {
  const float* x      = (const float*)d_in[0];  // (8,256,64,64)
  const float* w_conv = (const float*)d_in[1];  // (256,256)
  const float* bn1_g  = (const float*)d_in[2];  // (576,)
  const float* bn1_b  = (const float*)d_in[3];  // (576,)
  const float* w_key  = (const float*)d_in[4];  // (64,576)
  const float* bn2_g  = (const float*)d_in[5];  // (128,)
  const float* bn2_b  = (const float*)d_in[6];  // (128,)
  const float* w_cw   = (const float*)d_in[7];  // (768,128)
  float* out = (float*)d_out;                   // (8,128,64,64)

  // Workspace layout (~49.7 MB total, all fully written before read):
  float* T   = (float*)d_ws;                    // t:  8*256*4096 f32
  float* X2  = T  + (long)8 * 256 * HW;         // x2: 8*64*4096  f32
  float* QK  = X2 + (long)8 * 64  * HW;         // act:8*128*256  f32
  float* EMB = QK + (long)8 * 128 * 256;        // emb:8*768*256  f32
  unsigned short* WCB  = (unsigned short*)(EMB + (long)8 * 768 * 256); // bf16 w_conv
  unsigned short* WKB  = WCB + 256 * 256;                              // bf16 w_key
  unsigned short* WCWB = WKB + 64 * 576;                               // bf16 w_cw

  // Pre-convert weights to bf16 (tiny, one pass)
  k_cvt_bf16<<<dim3((65536 + 255) / 256), 256, 0, stream>>>(w_conv, WCB, 65536);
  k_cvt_bf16<<<dim3((36864 + 255) / 256), 256, 0, stream>>>(w_key, WKB, 36864);
  k_cvt_bf16<<<dim3((98304 + 255) / 256), 256, 0, stream>>>(w_cw, WCWB, 98304);

  // GEMM1: t[b] (256x4096) = w_conv (256x256) * x[b] (256x4096), K=256 -> 8 chunks
  k_gemm_bf16<<<dim3(32, 4, 8), 256, 0, stream>>>(
      WCB, 256, x, HW, (long)256 * HW, T, (long)256 * HW, 8);

  // GEMM2 (fused patch-gather + bn1 + relu): x2[b] (64x4096), K=576
  k_key_gemm<<<dim3(32, 1, 8), 256, 0, stream>>>(WKB, bn1_g, bn1_b, T, X2);

  // Pool + bn2 + relu
  k_pool_bn<<<dim3(1024), 256, 0, stream>>>(T, X2, bn2_g, bn2_b, QK);

  // GEMM3: embed[b] (768x256) = w_cw (768x128) * QK[b] (128x256), K=128 -> 4 chunks
  k_gemm_bf16<<<dim3(2, 12, 8), 256, 0, stream>>>(
      WCWB, 128, QK, 256, (long)128 * 256, EMB, (long)768 * 256, 4);

  // Separable 3x3 weighted sum
  k_apply<<<dim3(16384), 256, 0, stream>>>(EMB, T, out);
}
